// CovidCaps_91199335563674
// MI455X (gfx1250) — compile-verified, matrix operand order checked by reading, not run on hardware
//
#include <hip/hip_runtime.h>
#include <hip/hip_bf16.h>
#include <stdint.h>
#include <stddef.h>

typedef __attribute__((ext_vector_type(8)))  __bf16 v8bf;
typedef __attribute__((ext_vector_type(16))) __bf16 v16bf;
typedef __attribute__((ext_vector_type(8)))  float  v8f;

#define BN_EPS 0.001f
#define SQ_EPS 1e-7f

// ---------------------------------------------------------------------------
// conv1: 4x4x3 stride-2 direct conv (K=48, memory-streaming) + bias + ReLU +
// BN1, output stored bf16 (halves the dominant 467MB -> 234MB HBM tensor).
// ---------------------------------------------------------------------------
__global__ void conv1_kernel(const float* __restrict__ x,    // (32,480,480,3)
                             const float* __restrict__ w,    // (4,4,3,64)
                             const float* __restrict__ bias, // (64)
                             const float* __restrict__ g, const float* __restrict__ bb,
                             const float* __restrict__ mm, const float* __restrict__ vv,
                             __bf16* __restrict__ out)       // (32,239,239,64)
{
    size_t idx = (size_t)blockIdx.x * blockDim.x + threadIdx.x;
    const size_t total = (size_t)32 * 239 * 239 * 64;
    if (idx >= total) return;
    int c   = (int)(idx & 63);
    size_t p = idx >> 6;
    int wo = (int)(p % 239);
    int ho = (int)((p / 239) % 239);
    int b  = (int)(p / (239 * 239));
    int hi0 = ho * 2, wi0 = wo * 2;
    float acc = 0.f;
    #pragma unroll
    for (int kh = 0; kh < 4; ++kh) {
        #pragma unroll
        for (int kw = 0; kw < 4; ++kw) {
            const float* xp = x + (((size_t)b * 480 + (hi0 + kh)) * 480 + (wi0 + kw)) * 3;
            const float* wp = w + (size_t)((kh * 4 + kw) * 3) * 64 + c;
            acc += xp[0] * wp[0] + xp[1] * wp[64] + xp[2] * wp[128];
        }
    }
    acc += bias[c];
    acc = fmaxf(acc, 0.f);
    float sc = g[c] * rsqrtf(vv[c] + BN_EPS);
    acc = (acc - mm[c]) * sc + bb[c];
    out[idx] = (__bf16)acc;
}

// ---------------------------------------------------------------------------
// Pack 3x3x64xN conv weights (HWIO f32) into the WMMA B-matrix lane layout:
// dst[nt][j][lane][t], 16 contiguous bf16 per lane per 32-wide K chunk j.
// Per ISA 16-bit layout: lanes 0-15 hold chunk-K {0..7,16..23}, lanes 16-31
// hold {8..15,24..31}; vector elements are consecutive K within each group.
// ---------------------------------------------------------------------------
__global__ void pack_w_kernel(const float* __restrict__ w, // (3,3,64,N)
                              __bf16* __restrict__ dst, int N, int ntiles)
{
    int id = blockIdx.x * blockDim.x + threadIdx.x;
    int total = ntiles * 18 * 32 * 16;
    if (id >= total) return;
    int t    = id & 15;
    int lane = (id >> 4) & 31;
    int j    = (id >> 9) % 18;
    int nt   = id / (18 * 512);
    int base = (lane >= 16) ? 8 : 0;
    int kin  = (t < 8) ? (base + t) : (16 + base + (t - 8));
    int k    = j * 32 + kin;           // 0..575
    int tap  = k >> 6;                 // kh*3+kw (chunk never straddles a tap)
    int cc   = k & 63;                 // input channel
    int n    = nt * 16 + (lane & 15);  // output channel
    dst[id] = (__bf16)w[((size_t)tap * 64 + cc) * N + n];
}

__global__ void cvt_bf16_kernel(const float* __restrict__ src,
                                __bf16* __restrict__ dst, int n)
{
    int i = blockIdx.x * blockDim.x + threadIdx.x;
    if (i < n) dst[i] = (__bf16)src[i];
}

// ---------------------------------------------------------------------------
// Implicit-GEMM conv via WMMA bf16: 3x3, stride 2, Cin=64, register-blocked
// over ALL output-channel tiles (NT): one wave computes a 16(pixel) x NT*16
// (channel) strip -> A loaded once per K-chunk, NT back-to-back WMMAs.
// STAGE: packed weights are copied to LDS once per workgroup with the CDNA5
// async-copy path (GLOBAL_LOAD_ASYNC_TO_LDS_B128 / ASYNCcnt), so the B
// operand is read per-block from LDS (ds_load_b128) instead of per-wave from
// L2 (8x less B traffic). K=576 as 18 chunks of 32. Epilogue fuses bias +
// ReLU + optional BN, stores bf16 (lane predication only after the WMMAs).
// ---------------------------------------------------------------------------
template <int NT, bool STAGE>
__global__ void conv_wmma_kernel(const __bf16* __restrict__ in,    // (B,Hin,Win,64)
                                 const __bf16* __restrict__ wpack, // packed B
                                 const float* __restrict__ bias,   // (N)
                                 const float* __restrict__ g, const float* __restrict__ bb,
                                 const float* __restrict__ mm, const float* __restrict__ vv,
                                 int has_bn,
                                 __bf16* __restrict__ out,         // (B,P,N)
                                 int Hin, int Win, int Wout,
                                 int P, int Ptiles, int totalWaves)
{
    const int N = NT * 16;
    __shared__ __align__(32) __bf16 lds_w[STAGE ? (NT * 18 * 32 * 16) : 16];

    if constexpr (STAGE) {
        // Cooperative async copy of the whole packed-weight block into LDS.
        const int CHUNKS = NT * 18 * 32 * 2;       // 16-byte chunks
        uint32_t lbase = (uint32_t)(uintptr_t)(&lds_w[0]);
        for (int k = threadIdx.x; k < CHUNKS; k += 256) {
            uint32_t laddr = lbase + (uint32_t)k * 16u;
            uint32_t goff  = (uint32_t)k * 16u;    // byte offset from wpack
            asm volatile("global_load_async_to_lds_b128 %0, %1, %2"
                         :: "v"(laddr), "v"(goff), "s"(wpack)
                         : "memory");
        }
        asm volatile("s_wait_asynccnt 0x0" ::: "memory");
        __syncthreads();
    }

    int wid = (int)((blockIdx.x * blockDim.x + threadIdx.x) >> 5);
    if (wid >= totalWaves) return;     // wave-uniform: EXEC stays all-ones
    int lane = threadIdx.x & 31;
    int pt = wid % Ptiles;
    int b  = wid / Ptiles;
    int p0 = pt * 16;
    int hih  = lane >> 4;
    int base = hih * 8;
    int l15  = lane & 15;
    int pl = p0 + l15;                 // A-row pixel (clamped for loads)
    if (pl > P - 1) pl = P - 1;
    int ho = pl / Wout, wo = pl % Wout;

    v8f acc[NT] = {};
    #pragma unroll
    for (int j = 0; j < 18; ++j) {
        int tap = j >> 1;
        int kh = tap / 3, kw = tap % 3;
        int c0 = (j & 1) * 32;
        size_t off = (((size_t)b * Hin + (ho * 2 + kh)) * Win + (wo * 2 + kw)) * 64
                   + c0 + base;
        v8bf alo = *(const v8bf*)(in + off);
        v8bf ahi = *(const v8bf*)(in + off + 16);
        v16bf a = __builtin_shufflevector(alo, ahi,
                    0,1,2,3,4,5,6,7,8,9,10,11,12,13,14,15);
        #pragma unroll
        for (int nt = 0; nt < NT; ++nt) {
            size_t bidx = ((size_t)(nt * 18 + j) * 32 + lane) * 16;
            v16bf bm;
            if constexpr (STAGE) bm = *(const v16bf*)(lds_w + bidx);
            else                 bm = *(const v16bf*)(wpack + bidx);
            acc[nt] = __builtin_amdgcn_wmma_f32_16x16x32_bf16(
                          false, a, false, bm, (short)0, acc[nt], false, false);
        }
    }

    #pragma unroll
    for (int nt = 0; nt < NT; ++nt) {
        int n = nt * 16 + l15;
        float bi = bias[n];
        float sc = 1.f, sh = 0.f;
        if (has_bn) { sc = g[n] * rsqrtf(vv[n] + BN_EPS); sh = bb[n] - mm[n] * sc; }
        #pragma unroll
        for (int r = 0; r < 8; ++r) {
            int po = p0 + r + hih * 8; // D row m = r + 8*(lane>=16)
            if (po < P) {
                float v = acc[nt][r] + bi;
                v = fmaxf(v, 0.f);
                if (has_bn) v = v * sc + sh;
                out[((size_t)b * P + po) * N + n] = (__bf16)v;
            }
        }
    }
}

// ---------------------------------------------------------------------------
// 2x2 average pool, bf16 in/out.
// ---------------------------------------------------------------------------
__global__ void pool_kernel(const __bf16* __restrict__ in, __bf16* __restrict__ out,
                            int H, int W, int Ho, int Wo, int C, int total)
{
    int id = blockIdx.x * blockDim.x + threadIdx.x;
    if (id >= total) return;
    int c  = id % C;
    int r  = id / C;
    int wo = r % Wo;
    int ho = (r / Wo) % Ho;
    int b  = r / (Wo * Ho);
    size_t i00 = (((size_t)b * H + ho * 2) * W + wo * 2) * C + c;
    size_t rowS = (size_t)W * C;
    float s = (float)in[i00] + (float)in[i00 + C]
            + (float)in[i00 + rowS] + (float)in[i00 + rowS + C];
    out[id] = (__bf16)(0.25f * s);
}

// ---------------------------------------------------------------------------
// u_hat[b,j,i,d] = sum_k W[j,i,d,k] * u[b,i,k]  via WMMA. One wave per (j,i)
// pair; the W-matrix A operand is loaded once per K-chunk and reused for both
// 16-batch halves (2 accumulators, 8 WMMA per wave, 180 waves).
// ---------------------------------------------------------------------------
__global__ void uhat_wmma_kernel(const __bf16* __restrict__ Wc,  // (5,36,16,128)
                                 const __bf16* __restrict__ u,   // (32,36,128)
                                 float* __restrict__ uhat)       // (32,5,36,16)
{
    int wid = (int)((blockIdx.x * blockDim.x + threadIdx.x) >> 5);
    if (wid >= 180) return;
    int lane = threadIdx.x & 31;
    int i  = wid % 36;
    int j  = wid / 36;
    int hih = lane >> 4, base = hih * 8;
    int l15 = lane & 15;

    v8f acc[2] = {};
    #pragma unroll
    for (int q = 0; q < 4; ++q) {
        size_t aoff = ((size_t)(j * 36 + i) * 16 + l15) * 128 + q * 32 + base;
        v8bf alo = *(const v8bf*)(Wc + aoff);
        v8bf ahi = *(const v8bf*)(Wc + aoff + 16);
        v16bf a = __builtin_shufflevector(alo, ahi,
                    0,1,2,3,4,5,6,7,8,9,10,11,12,13,14,15);
        #pragma unroll
        for (int bt = 0; bt < 2; ++bt) {
            size_t boff = ((size_t)(bt * 16 + l15) * 36 + i) * 128 + q * 32 + base;
            v8bf blo = *(const v8bf*)(u + boff);
            v8bf bhi = *(const v8bf*)(u + boff + 16);
            v16bf bm = __builtin_shufflevector(blo, bhi,
                        0,1,2,3,4,5,6,7,8,9,10,11,12,13,14,15);
            acc[bt] = __builtin_amdgcn_wmma_f32_16x16x32_bf16(
                          false, a, false, bm, (short)0, acc[bt], false, false);
        }
    }
    #pragma unroll
    for (int bt = 0; bt < 2; ++bt) {
        int bb = bt * 16 + l15;        // D column n = batch
        #pragma unroll
        for (int r = 0; r < 8; ++r) {
            int d = r + hih * 8;       // D row m = capsule dim
            uhat[(((size_t)bb * 5 + j) * 36 + i) * 16 + d] = acc[bt][r];
        }
    }
}

// ---------------------------------------------------------------------------
// Dynamic routing (4 iters) + final norms. One block per batch element,
// everything in LDS (5*36*16 floats = 11.25KB).
// ---------------------------------------------------------------------------
__global__ void routing_kernel(const float* __restrict__ uhat,   // (32,5,36,16)
                               float* __restrict__ outn)         // (32,5)
{
    __shared__ float uh[5 * 36 * 16];
    __shared__ float bl[5][36];
    __shared__ float cc[5][36];
    __shared__ float sj[5][16];
    __shared__ float vvv[5][16];
    __shared__ float scl[5];
    int b = blockIdx.x;
    int t = threadIdx.x;
    for (int k = t; k < 2880; k += 192) uh[k] = uhat[(size_t)b * 2880 + k];
    if (t < 180) bl[t / 36][t % 36] = 0.f;
    __syncthreads();
    for (int it = 0; it < 4; ++it) {
        if (t < 36) {                              // softmax over j per i
            float mx = bl[0][t];
            for (int j = 1; j < 5; ++j) mx = fmaxf(mx, bl[j][t]);
            float e[5]; float s = 0.f;
            for (int j = 0; j < 5; ++j) { e[j] = __expf(bl[j][t] - mx); s += e[j]; }
            for (int j = 0; j < 5; ++j) cc[j][t] = e[j] / s;
        }
        __syncthreads();
        if (t < 80) {                              // s[j][d] = sum_i c*u_hat
            int j = t >> 4, d = t & 15;
            float a = 0.f;
            for (int i = 0; i < 36; ++i) a += cc[j][i] * uh[(j * 36 + i) * 16 + d];
            sj[j][d] = a;
        }
        __syncthreads();
        if (t < 5) {                               // squash scale
            float s2 = SQ_EPS;
            for (int d = 0; d < 16; ++d) s2 += sj[t][d] * sj[t][d];
            scl[t] = sqrtf(s2) / (1.f + s2);
        }
        __syncthreads();
        if (t < 80) { int j = t >> 4, d = t & 15; vvv[j][d] = scl[j] * sj[j][d]; }
        __syncthreads();
        if (it < 3 && t < 180) {                   // b += <out, u_hat>
            int j = t / 36, i = t % 36;
            float a = 0.f;
            for (int d = 0; d < 16; ++d) a += vvv[j][d] * uh[(j * 36 + i) * 16 + d];
            bl[j][i] += a;
        }
        __syncthreads();
    }
    if (t < 5) {
        float s = 0.f;
        for (int d = 0; d < 16; ++d) s += vvv[t][d] * vvv[t][d];
        outn[b * 5 + t] = sqrtf(s);
    }
}

// ---------------------------------------------------------------------------
static inline unsigned cdiv(size_t a, unsigned b) { return (unsigned)((a + b - 1) / b); }

extern "C" void kernel_launch(void* const* d_in, const int* in_sizes, int n_in,
                              void* d_out, int out_size, void* d_ws, size_t ws_size,
                              hipStream_t stream)
{
    (void)in_sizes; (void)n_in; (void)out_size; (void)ws_size;
    const float* x    = (const float*)d_in[0];
    const float* w1   = (const float*)d_in[1];
    const float* b1   = (const float*)d_in[2];
    const float* bn1g = (const float*)d_in[3];
    const float* bn1b = (const float*)d_in[4];
    const float* bn1m = (const float*)d_in[5];
    const float* bn1v = (const float*)d_in[6];
    const float* w2   = (const float*)d_in[7];
    const float* b2   = (const float*)d_in[8];
    const float* w3   = (const float*)d_in[9];
    const float* b3   = (const float*)d_in[10];
    const float* bn2g = (const float*)d_in[11];
    const float* bn2b = (const float*)d_in[12];
    const float* bn2m = (const float*)d_in[13];
    const float* bn2v = (const float*)d_in[14];
    const float* w4   = (const float*)d_in[15];
    const float* b4   = (const float*)d_in[16];
    const float* bn3g = (const float*)d_in[17];
    const float* bn3b = (const float*)d_in[18];
    const float* bn3m = (const float*)d_in[19];
    const float* bn3v = (const float*)d_in[20];
    const float* cW   = (const float*)d_in[21];
    float* out = (float*)d_out;

    // ---- workspace carve-out (256B aligned) ----
    char* ws = (char*)d_ws;
    size_t off = 0;
    auto carve = [&](size_t bytes) -> char* {
        char* p = ws + off;
        off += (bytes + 255) & ~(size_t)255;
        return p;
    };
    const size_t nY1 = (size_t)32 * 239 * 239 * 64;
    const size_t nY2 = (size_t)32 * 119 * 119 * 64;
    const size_t nY2p = (size_t)32 * 59 * 59 * 64;
    const size_t nY3 = (size_t)32 * 29 * 29 * 64;
    const size_t nY3p = (size_t)32 * 14 * 14 * 64;
    const size_t nU  = (size_t)32 * 36 * 128;
    __bf16* y1b  = (__bf16*)carve(nY1 * 2);
    __bf16* y2b  = (__bf16*)carve(nY2 * 2);
    __bf16* y2p  = (__bf16*)carve(nY2p * 2);
    __bf16* y3b  = (__bf16*)carve(nY3 * 2);
    __bf16* y3p  = (__bf16*)carve(nY3p * 2);
    __bf16* ub   = (__bf16*)carve(nU * 2);
    __bf16* wp2  = (__bf16*)carve((size_t)4 * 18 * 32 * 16 * 2);
    __bf16* wp3  = (__bf16*)carve((size_t)4 * 18 * 32 * 16 * 2);
    __bf16* wp4  = (__bf16*)carve((size_t)8 * 18 * 32 * 16 * 2);
    __bf16* cWb  = (__bf16*)carve((size_t)5 * 36 * 16 * 128 * 2);
    float*  uhat = (float*)carve((size_t)32 * 5 * 36 * 16 * 4);

    // ---- 1) conv1 + ReLU + BN1 -> bf16 ----
    conv1_kernel<<<cdiv(nY1, 256), 256, 0, stream>>>(
        x, w1, b1, bn1g, bn1b, bn1m, bn1v, y1b);

    // ---- 2) weight packing / conversions ----
    pack_w_kernel<<<cdiv((size_t)4 * 18 * 32 * 16, 256), 256, 0, stream>>>(w2, wp2, 64, 4);
    pack_w_kernel<<<cdiv((size_t)4 * 18 * 32 * 16, 256), 256, 0, stream>>>(w3, wp3, 64, 4);
    pack_w_kernel<<<cdiv((size_t)8 * 18 * 32 * 16, 256), 256, 0, stream>>>(w4, wp4, 128, 8);
    cvt_bf16_kernel<<<cdiv((size_t)5 * 36 * 16 * 128, 256), 256, 0, stream>>>(
        cW, cWb, 5 * 36 * 16 * 128);

    // ---- 3) conv2 (WMMA, NT=4, LDS-staged B) + ReLU -> y2b ----
    {
        int P = 119 * 119, Pt = (P + 15) / 16;
        int waves = 32 * Pt;
        conv_wmma_kernel<4, true><<<cdiv((size_t)waves * 32, 256), 256, 0, stream>>>(
            y1b, wp2, b2, nullptr, nullptr, nullptr, nullptr, 0,
            y2b, 239, 239, 119, P, Pt, waves);
    }
    // ---- 4) avgpool 119 -> 59 ----
    pool_kernel<<<cdiv(nY2p, 256), 256, 0, stream>>>(
        y2b, y2p, 119, 119, 59, 59, 64, (int)nY2p);

    // ---- 5) conv3 (WMMA, NT=4, LDS-staged B) + ReLU + BN2 -> y3b ----
    {
        int P = 29 * 29, Pt = (P + 15) / 16;
        int waves = 32 * Pt;
        conv_wmma_kernel<4, true><<<cdiv((size_t)waves * 32, 256), 256, 0, stream>>>(
            y2p, wp3, b3, bn2g, bn2b, bn2m, bn2v, 1,
            y3b, 59, 59, 29, P, Pt, waves);
    }
    // ---- 6) avgpool 29 -> 14 ----
    pool_kernel<<<cdiv(nY3p, 256), 256, 0, stream>>>(
        y3b, y3p, 29, 29, 14, 14, 64, (int)nY3p);

    // ---- 7) conv4 (WMMA, NT=8, direct-global B: only 96 waves) ----
    {
        int P = 6 * 6, Pt = (P + 15) / 16;
        int waves = 32 * Pt;
        conv_wmma_kernel<8, false><<<cdiv((size_t)waves * 32, 256), 256, 0, stream>>>(
            y3p, wp4, b4, bn3g, bn3b, bn3m, bn3v, 1,
            ub, 14, 14, 6, P, Pt, waves);
    }

    // ---- 8) u_hat (WMMA), 180 waves ----
    uhat_wmma_kernel<<<cdiv((size_t)180 * 32, 256), 256, 0, stream>>>(cWb, ub, uhat);

    // ---- 9) routing + norms ----
    routing_kernel<<<32, 192, 0, stream>>>(uhat, out);
}
